// SAttention_fast_63410897158188
// MI455X (gfx1250) — compile-verified, matrix operand order checked by reading, not run on hardware
//
#include <hip/hip_runtime.h>
#include <hip/hip_bf16.h>

#define N_TRACKS 1024
#define EDIM 32
#define HDIM 256
#define GATE 128
#define THREADS 256
#define NWAVES 8

typedef __attribute__((ext_vector_type(16))) _Float16 v16h;
typedef __attribute__((ext_vector_type(8)))  float    v8f;

// fast sigmoid/tanh: v_exp_f32 + v_rcp_f32 (no IEEE div sequences)
__device__ __forceinline__ float fsig(float x) {
    return __builtin_amdgcn_rcpf(1.0f + __expf(-x));
}
__device__ __forceinline__ float ftanhf(float x) {
    float cx = fminf(fmaxf(x, -15.f), 15.f);
    float e = __expf(2.f * cx);
    return (e - 1.f) * __builtin_amdgcn_rcpf(e + 1.f);
}

__global__ __launch_bounds__(THREADS, 1)
void sattn_fused_kernel(const float* __restrict__ hidden_state,
                        const float* __restrict__ obs2,
                        const float* __restrict__ h0,
                        const float* __restrict__ c0,
                        const float* __restrict__ emb_w,  const float* __restrict__ emb_b,
                        const float* __restrict__ hts_w,  const float* __restrict__ hts_b,
                        const float* __restrict__ w_ih,   const float* __restrict__ b_ih,
                        const float* __restrict__ w_hh,   const float* __restrict__ b_hh,
                        const float* __restrict__ wq,     const float* __restrict__ wk,
                        const float* __restrict__ wv,
                        const float* __restrict__ in_proj_w, const float* __restrict__ in_proj_b,
                        const float* __restrict__ attn_out_w, const float* __restrict__ attn_out_b,
                        const float* __restrict__ proj_w, const float* __restrict__ proj_b,
                        float* __restrict__ out)
{
    __shared__ __align__(32) _Float16 sB[16 * 32 * 16]; // 16 frags x 32 lanes x 16 halfs = 16KB
    __shared__ __align__(16) float4 sEmbP[EDIM];        // (w0, w1, b, 0) per k
    __shared__ float sBias[GATE];
    __shared__ float sKq[EDIM];
    __shared__ float sHs[EDIM];
    __shared__ float sS0Kb[2];
    __shared__ float sWaveM[NWAVES], sWaveL[NWAVES];
    __shared__ float sWaveAcc[NWAVES][EDIM];

    const int b    = blockIdx.x;
    const int tid  = threadIdx.x;
    const int lane = tid & 31;
    const int wave = tid >> 5;
    const int n    = lane & 15;   // tile column / A-row-in-chunk for this lane
    const int hi   = lane >> 4;   // wave half: 0 -> K 0..7/16..23, rows 0..7 ; 1 -> K 8..15/24..31, rows 8..15

    // ---- pack WMMA B fragments (f16) for w_ih / w_hh into LDS ----
    // frag f = t*2 + kh ; t = output tile (gate cols 16t..16t+15); kh: 0 = x-part(w_ih), 1 = h-part(w_hh)
    for (int idx = tid; idx < 16 * 32; idx += THREADS) {
        const int f = idx >> 5, L = idx & 31;
        const int t = f >> 1, kh = f & 1;
        const int col = (t << 4) + (L & 15);
        const int kb0 = (L < 16) ? 0 : 8;
        const float* W = kh ? w_hh : w_ih;
        const float* row = W + col * EDIM;
        _Float16* dst = &sB[idx * 16];
        #pragma unroll
        for (int h = 0; h < 16; ++h) {
            const int k = kb0 + ((h < 8) ? h : (h + 8));
            dst[h] = (_Float16)row[k];
        }
    }
    if (tid < GATE) sBias[tid] = b_ih[tid] + b_hh[tid];
    if (tid < EDIM) sEmbP[tid] = make_float4(emb_w[2 * tid], emb_w[2 * tid + 1], emb_b[tid], 0.f);

    // ---- wave 0: hs[b], q0, kq (scaled), kb, s0 via shuffle matvecs ----
    if (wave == 0) {
        float hs = hts_b[lane];
        const float* hr = hidden_state + (size_t)b * HDIM;
        const float* wr = hts_w + lane * HDIM;
        for (int k = 0; k < HDIM; ++k) hs = fmaf(hr[k], wr[k], hs);
        float tq = 0.f;
        for (int k = 0; k < EDIM; ++k) tq = fmaf(wq[lane * EDIM + k], __shfl(hs, k, 32), tq);
        float q0 = in_proj_b[lane];
        for (int j = 0; j < EDIM; ++j) q0 = fmaf(in_proj_w[lane * EDIM + j], __shfl(tq, j, 32), q0);
        float u = 0.f;
        for (int l = 0; l < EDIM; ++l) u = fmaf(in_proj_w[(EDIM + l) * EDIM + lane], __shfl(q0, l, 32), u);
        float kq = 0.f;
        for (int j = 0; j < EDIM; ++j) kq = fmaf(wk[j * EDIM + lane], __shfl(u, j, 32), kq);
        float kb = in_proj_b[EDIM + lane] * q0;
        for (int m = 16; m >= 1; m >>= 1) kb += __shfl_xor(kb, m, 32);
        const float sc = 0.17677669529663687f; // 1/sqrt(32)
        kq *= sc; kb *= sc;
        sKq[lane] = kq; sHs[lane] = hs;
        float s0 = hs * kq;
        for (int m = 16; m >= 1; m >>= 1) s0 += __shfl_xor(s0, m, 32);
        s0 += kb;
        if (lane == 0) { sS0Kb[0] = s0; sS0Kb[1] = kb; }
    }
    __syncthreads();

    // ---- hoist only the h-part (w_hh) B fragments into registers (64 VGPRs);
    //      x-part fragments are streamed from LDS inside the loop ----
    v16h Bh[8];
    #pragma unroll
    for (int t = 0; t < 8; ++t) Bh[t] = *(const v16h*)&sB[((t * 2 + 1) * 32 + lane) * 16];

    // loop-invariant per-lane bias values: bias[gate][ch] with e = n + 16*ch
    float bias_r[4][2];
    #pragma unroll
    for (int g = 0; g < 4; ++g) {
        bias_r[g][0] = sBias[g * 32 + n];
        bias_r[g][1] = sBias[g * 32 + n + 16];
    }

    const float kb  = sS0Kb[1];
    const float kq0 = sKq[n];
    const float kq1 = sKq[n + 16];
    const float ob0 = obs2[2 * b], ob1 = obs2[2 * b + 1];

    float m_run = -3.0e38f, l_run = 0.f, acc0 = 0.f, acc1 = 0.f;
    const int mrow_base = hi * 8;
    const int kbase = hi * 8;

    // wave-uniform bases (stay in SGPRs); hot-loop indices are 32-bit offsets
    const float*  hbase  = h0 + (size_t)b * N_TRACKS * EDIM;
    const float4* hbase4 = (const float4*)hbase;
    const float*  cbase  = c0 + (size_t)b * N_TRACKS * EDIM;

    // ---- main loop: 8 chunks of 16 neighbors per wave ----
    for (int cc = wave; cc < 64; cc += NWAVES) {
        const int qbase = cc * 16;
        const int qc = min(qbase + n, 1022);
        const int j  = (qc < b) ? qc : qc + 1;

        // prefetch next chunk's h0/c0 rows
        {
            const int cn = cc + NWAVES;
            if (cn < 64) {
                const int qn = min(cn * 16 + n, 1022);
                const int jn = (qn < b) ? qn : qn + 1;
                __builtin_prefetch(hbase + jn * EDIM, 0, 3);
                __builtin_prefetch(cbase + jn * EDIM, 0, 3);
            }
        }

        const float r0 = obs2[2 * j]     - ob0;
        const float r1 = obs2[2 * j + 1] - ob1;

        const int hoff = j * (EDIM / 4); // float4 units
        const float4 hA0 = hbase4[hoff + (kbase >> 2) + 0];
        const float4 hA1 = hbase4[hoff + (kbase >> 2) + 1];
        const float4 hB0 = hbase4[hoff + ((kbase + 16) >> 2) + 0];
        const float4 hB1 = hbase4[hoff + ((kbase + 16) >> 2) + 1];

        v16h ax, ah;
        ah[0]  = (_Float16)hA0.x; ah[1]  = (_Float16)hA0.y; ah[2]  = (_Float16)hA0.z; ah[3]  = (_Float16)hA0.w;
        ah[4]  = (_Float16)hA1.x; ah[5]  = (_Float16)hA1.y; ah[6]  = (_Float16)hA1.z; ah[7]  = (_Float16)hA1.w;
        ah[8]  = (_Float16)hB0.x; ah[9]  = (_Float16)hB0.y; ah[10] = (_Float16)hB0.z; ah[11] = (_Float16)hB0.w;
        ah[12] = (_Float16)hB1.x; ah[13] = (_Float16)hB1.y; ah[14] = (_Float16)hB1.z; ah[15] = (_Float16)hB1.w;
        #pragma unroll
        for (int h = 0; h < 16; ++h) {
            const int k = kbase + ((h < 8) ? h : (h + 8));
            const float4 ep = sEmbP[k];
            const float xv = fmaxf(fmaf(r0, ep.x, fmaf(r1, ep.y, ep.z)), 0.f);
            ax[h] = (_Float16)xv;
        }

        // per-row c0 element offsets (32-bit), shared by both column-half passes
        int coff[8];
        #pragma unroll
        for (int r = 0; r < 8; ++r) {
            const int qm = min(qbase + mrow_base + r, 1022);
            const int jm = (qm < b) ? qm : qm + 1;
            coff[r] = jm * EDIM;
        }

        // two column-half passes (tiles {0,2,4,6} for e=n ; {1,3,5,7} for e=n+16)
        float hn0[8], hn1[8];
        #pragma unroll
        for (int ch = 0; ch < 2; ++ch) {
            v8f accT[4];
            #pragma unroll
            for (int g = 0; g < 4; ++g) {
                const int t = 2 * g + ch;
                const v16h bx = *(const v16h*)&sB[((t * 2) * 32 + lane) * 16];
                v8f z = {0.f, 0.f, 0.f, 0.f, 0.f, 0.f, 0.f, 0.f};
                v8f c1 = __builtin_amdgcn_wmma_f32_16x16x32_f16(false, ax, false, bx,
                                                                (short)0, z, false, false);
                accT[g] = __builtin_amdgcn_wmma_f32_16x16x32_f16(false, ah, false, Bh[t],
                                                                 (short)0, c1, false, false);
            }
            const int e = n + 16 * ch;
            #pragma unroll
            for (int r = 0; r < 8; ++r) {
                const float cv = cbase[coff[r] + e];
                const float iv = accT[0][r] + bias_r[0][ch];
                const float fv = accT[1][r] + bias_r[1][ch];
                const float gv = accT[2][r] + bias_r[2][ch];
                const float ov = accT[3][r] + bias_r[3][ch];
                const float cn = fsig(fv) * cv + fsig(iv) * ftanhf(gv);
                const float hn = fsig(ov) * ftanhf(cn);
                if (ch == 0) hn0[r] = hn; else hn1[r] = hn;
            }
        }

        // scores: dot(h_new, kq) reduced within each 16-lane half
        float scr[8];
        #pragma unroll
        for (int r = 0; r < 8; ++r) {
            float part = hn0[r] * kq0 + hn1[r] * kq1;
            part += __shfl_xor(part, 1, 32);
            part += __shfl_xor(part, 2, 32);
            part += __shfl_xor(part, 4, 32);
            part += __shfl_xor(part, 8, 32);
            scr[r] = (qbase + mrow_base + r < 1023) ? (part + kb) : -3.0e38f;
        }

        // online softmax update
        float cmax = scr[0];
        #pragma unroll
        for (int r = 1; r < 8; ++r) cmax = fmaxf(cmax, scr[r]);
        cmax = fmaxf(cmax, __shfl_xor(cmax, 16, 32));
        const float m_new = fmaxf(m_run, cmax);
        const float scale = __expf(m_run - m_new);
        l_run *= scale; acc0 *= scale; acc1 *= scale;
        #pragma unroll
        for (int r = 0; r < 8; ++r) {
            const float w = __expf(scr[r] - m_new);
            l_run += w;
            acc0 = fmaf(w, hn0[r], acc0);
            acc1 = fmaf(w, hn1[r], acc1);
        }
        m_run = m_new;
    }

    // ---- per-wave combine (halves share m_run) ----
    const float accO0 = acc0 + __shfl_xor(acc0, 16, 32);
    const float accO1 = acc1 + __shfl_xor(acc1, 16, 32);
    const float lT    = l_run + __shfl_xor(l_run, 16, 32);
    if (lane < 16) { sWaveAcc[wave][lane] = accO0; sWaveAcc[wave][lane + 16] = accO1; }
    if (lane == 0) { sWaveM[wave] = m_run; sWaveL[wave] = lT; }
    __syncthreads();

    // ---- final combine + output projections (wave 0) ----
    if (wave == 0) {
        const float s0 = sS0Kb[0];
        float mg = s0;
        #pragma unroll
        for (int w = 0; w < NWAVES; ++w) mg = fmaxf(mg, sWaveM[w]);
        float Lg   = __expf(s0 - mg);
        float accE = Lg * sHs[lane];
        #pragma unroll
        for (int w = 0; w < NWAVES; ++w) {
            const float f = __expf(sWaveM[w] - mg);
            Lg  += f * sWaveL[w];
            accE = fmaf(f, sWaveAcc[w][lane], accE);
        }
        const float avg = accE * __builtin_amdgcn_rcpf(Lg);
        // ctx = wv_in @ (wv @ avg) + bv ; then attn_out, proj
        float t1 = 0.f;
        for (int k = 0; k < EDIM; ++k) t1 = fmaf(wv[lane * EDIM + k], __shfl(avg, k, 32), t1);
        float ctx = in_proj_b[2 * EDIM + lane];
        for (int j = 0; j < EDIM; ++j) ctx = fmaf(in_proj_w[(2 * EDIM + lane) * EDIM + j], __shfl(t1, j, 32), ctx);
        float av = attn_out_b[lane];
        for (int j = 0; j < EDIM; ++j) av = fmaf(attn_out_w[lane * EDIM + j], __shfl(ctx, j, 32), av);
        float o = proj_b[lane];
        for (int j = 0; j < EDIM; ++j) o = fmaf(proj_w[lane * EDIM + j], __shfl(av, j, 32), o);
        out[(size_t)b * EDIM + lane] = o;
    }
}

extern "C" void kernel_launch(void* const* d_in, const int* in_sizes, int n_in,
                              void* d_out, int out_size, void* d_ws, size_t ws_size,
                              hipStream_t stream) {
    (void)in_sizes; (void)n_in; (void)d_ws; (void)ws_size; (void)out_size;
    const float* hidden_state = (const float*)d_in[0];
    // d_in[1] = obs1 (unused by reference)
    const float* obs2       = (const float*)d_in[2];
    const float* h0         = (const float*)d_in[3];
    const float* c0         = (const float*)d_in[4];
    const float* emb_w      = (const float*)d_in[5];
    const float* emb_b      = (const float*)d_in[6];
    const float* hts_w      = (const float*)d_in[7];
    const float* hts_b      = (const float*)d_in[8];
    const float* w_ih       = (const float*)d_in[9];
    const float* b_ih       = (const float*)d_in[10];
    const float* w_hh       = (const float*)d_in[11];
    const float* b_hh       = (const float*)d_in[12];
    const float* wq         = (const float*)d_in[13];
    const float* wk         = (const float*)d_in[14];
    const float* wv         = (const float*)d_in[15];
    const float* in_proj_w  = (const float*)d_in[16];
    const float* in_proj_b  = (const float*)d_in[17];
    const float* attn_out_w = (const float*)d_in[18];
    const float* attn_out_b = (const float*)d_in[19];
    const float* proj_w     = (const float*)d_in[20];
    const float* proj_b     = (const float*)d_in[21];
    float* out = (float*)d_out;

    sattn_fused_kernel<<<dim3(N_TRACKS), dim3(THREADS), 0, stream>>>(
        hidden_state, obs2, h0, c0, emb_w, emb_b, hts_w, hts_b,
        w_ih, b_ih, w_hh, b_hh, wq, wk, wv, in_proj_w, in_proj_b,
        attn_out_w, attn_out_b, proj_w, proj_b, out);
}
